// ExodusNetwork_10316511445304
// MI455X (gfx1250) — matrix operand order repaired and checked by reading, not compiled, not run on hardware
//
#include <hip/hip_runtime.h>
#include <hip/hip_bf16.h>
#include <math.h>

typedef _Float16 h16;
typedef __attribute__((ext_vector_type(2)))  _Float16 v2h;
typedef __attribute__((ext_vector_type(8)))  _Float16 v8h;
typedef __attribute__((ext_vector_type(16))) _Float16 v16h;
typedef __attribute__((ext_vector_type(8)))  float    v8f;

#define THRESH 0.1f

// ---------------------------------------------------------------------------
// gfx1250 async global->LDS copy (16 bytes per lane), ASYNCcnt-tracked.
// vdst = VGPR holding LDS byte offset, vaddr = 64-bit global address.
// ---------------------------------------------------------------------------
__device__ __forceinline__ void async_copy_b128(unsigned lds_off, const void* gptr) {
  asm volatile("global_load_async_to_lds_b128 %0, %1, off"
               :: "v"(lds_off), "v"((unsigned long long)(size_t)gptr)
               : "memory");
}
__device__ __forceinline__ void wait_async_zero() {
  asm volatile("s_wait_asynccnt 0x0" ::: "memory");
}

// ---------------------------------------------------------------------------
// Weight norm + pack: w = g*v/||v||, stored f16 in B-fragment-friendly layout:
//   wB[((k>>4)*COUTP + co)*16 + (k&15)]
// so a lane's 16 B-halves (fixed column, k-group of 16) are 32 contiguous bytes.
// ---------------------------------------------------------------------------
template<int CIN, int COUT>
__global__ void conv_wnorm_kernel(const float* __restrict__ v,
                                  const float* __restrict__ g,
                                  h16* __restrict__ wB) {
  constexpr int K     = CIN * 9;
  constexpr int KP    = ((K + 31) / 32) * 32;
  constexpr int COUTP = ((COUT + 15) / 16) * 16;
  const int co  = blockIdx.x;   // 0..COUTP-1
  const int tid = threadIdx.x;  // 64
  __shared__ float red[64];
  float scale = 0.0f;
  if (co < COUT) {
    float s = 0.f;
    for (int k = tid; k < K; k += 64) {
      float w = v[co * K + k];
      s += w * w;
    }
    red[tid] = s;
    __syncthreads();
    for (int off = 32; off > 0; off >>= 1) {
      if (tid < off) red[tid] += red[tid + off];
      __syncthreads();
    }
    scale = g[co] / sqrtf(red[0]);
  }
  for (int k = tid; k < KP; k += 64) {
    float w = (co < COUT && k < K) ? v[co * K + k] * scale : 0.f;
    wB[(size_t)((k >> 4) * COUTP + co) * 16 + (k & 15)] = (h16)w;
  }
}

// Linear weight norm: (11,1024) -> f16 packed [k/16][16][k%16], cols 11..15 zero.
__global__ void lin_wnorm_kernel(const float* __restrict__ v,
                                 const float* __restrict__ g,
                                 h16* __restrict__ wB) {
  constexpr int K = 1024;
  const int co  = blockIdx.x;   // 0..15
  const int tid = threadIdx.x;  // 256
  __shared__ float red[256];
  float scale = 0.f;
  if (co < 11) {
    float s = 0.f;
    for (int k = tid; k < K; k += 256) { float w = v[co * K + k]; s += w * w; }
    red[tid] = s;
    __syncthreads();
    for (int off = 128; off > 0; off >>= 1) {
      if (tid < off) red[tid] += red[tid + off];
      __syncthreads();
    }
    scale = g[co] / sqrtf(red[0]);
  }
  for (int k = tid; k < K; k += 256)
    wB[(size_t)((k >> 4) * 16 + co) * 16 + (k & 15)] =
        (co < 11) ? (h16)(v[co * K + k] * scale) : (h16)0.f;
}

// ---------------------------------------------------------------------------
// Implicit-GEMM 3x3 SAME conv via V_WMMA_F32_16X16X32_F16.
// One block = one image n, 64 spatial positions (4 waves x 16 M-rows).
// Stage 1: input patch (rows with halo) -> LDS f16
//          (async global->LDS b128 for f16 inputs; manual cvt path for f32).
// Stage 2: im2col A-tile [64][KP] -> LDS (padding/bounds resolved here).
// Inner loop: A = 2x ds_load_b128, B = 2x global b128, 1 wmma per K-chunk.
// C store: 8 consecutive positions per lane -> one 16-byte store.
// ---------------------------------------------------------------------------
template<int CIN, int COUT, int S, typename TIN>
__global__ __launch_bounds__(128)
void conv_wmma_kernel(const TIN* __restrict__ in, const h16* __restrict__ wB,
                      h16* __restrict__ syn) {
  constexpr int K     = CIN * 9;
  constexpr int KP    = ((K + 31) / 32) * 32;
  constexpr int KC    = KP / 32;
  constexpr int COUTP = ((COUT + 15) / 16) * 16;
  constexpr int NT    = COUTP / 16;
  constexpr int ROWS  = 64 / S;       // image rows covered by 64 positions
  constexpr int RH    = ROWS + 2;     // + vertical halo
  __shared__ __align__(16) h16 patch[CIN * RH * S];
  __shared__ __align__(16) h16 atile[64 * KP];

  const int n    = blockIdx.x;
  const int m0b  = blockIdx.y * 64;
  const int y0   = m0b / S;
  const int tid  = threadIdx.x;
  const int wave = tid >> 5;
  const int lane = tid & 31;
  const int mrow = lane & 15;
  const int hi   = lane >> 4;

  // ---- Stage 1: input patch (rows y0-1 .. y0+ROWS) for all CIN, f16.
  const size_t inBase = (size_t)n * CIN * S * S;
  if constexpr (sizeof(TIN) == 2) {
    // f16 input: async DMA 16-byte chunks straight into LDS; zero-fill halo.
    constexpr int CPR    = S / 8;             // 16B chunks per row (S >= 8)
    constexpr int CHUNKS = CIN * RH * CPR;
    const unsigned ldsBase = (unsigned)(size_t)(void*)&patch[0];
    for (int i = tid; i < CHUNKS; i += 128) {
      int cin = i / (RH * CPR);
      int rem = i - cin * (RH * CPR);
      int ry  = rem / CPR;
      int xc  = rem - ry * CPR;
      int gy  = y0 - 1 + ry;
      int po  = (cin * RH + ry) * S + xc * 8;  // halves into patch
      if (gy >= 0 && gy < S) {
        const void* src = (const void*)(in + inBase + (size_t)cin * S * S +
                                        (size_t)gy * S + xc * 8);
        async_copy_b128(ldsBase + (unsigned)(po * 2), src);
      } else {
        v8h z = {};
        *(v8h*)(&patch[po]) = z;
      }
    }
    wait_async_zero();
  } else {
    // f32 input (layer 0): load + convert.
    for (int i = tid; i < CIN * RH * S; i += 128) {
      int cin = i / (RH * S);
      int rem = i - cin * (RH * S);
      int ry  = rem / S;
      int x   = rem - ry * S;
      int gy  = y0 - 1 + ry;
      float val = (gy >= 0 && gy < S)
                    ? (float)in[inBase + (size_t)cin * S * S + gy * S + x]
                    : 0.f;
      patch[i] = (h16)val;
    }
  }
  __syncthreads();

  // ---- Stage 2: im2col A-tile [m][k] (k-major across threads).
  for (int i = tid; i < 64 * KP; i += 128) {
    int m   = i / KP;
    int k   = i - m * KP;
    int pos = m0b + m;
    int py  = pos / S;
    int px  = pos - py * S;
    h16 val = (h16)0.f;
    if (k < K) {
      int cin = k / 9;
      int r   = k - cin * 9;
      int dy  = r / 3 - 1;
      int dx  = r - (dy + 1) * 3 - 1;
      int yy  = py + dy;
      int xx  = px + dx;
      if (xx >= 0 && xx < S) {
        int ry = yy - (y0 - 1);
        val = patch[cin * RH * S + ry * S + xx];
      }
    }
    atile[i] = val;
  }
  __syncthreads();

  const int m0 = m0b + wave * 16;
  const h16* arow = &atile[mrow * KP];

#pragma unroll
  for (int nt = 0; nt < NT; ++nt) {
    v8f c = {};
#pragma unroll
    for (int kc = 0; kc < KC; ++kc) {
      // A fragment: two contiguous 16B runs per lane.
      v8h a0 = *(const v8h*)(arow + kc * 32 + hi * 16);
      v8h a1 = *(const v8h*)(arow + kc * 32 + hi * 16 + 8);
      v16h a;
#pragma unroll
      for (int i = 0; i < 8; ++i) { a[i] = a0[i]; a[8 + i] = a1[i]; }
      // B fragment: 32 contiguous bytes (k-group = kc*2+hi, fixed column).
      const h16* bp = wB + (size_t)((kc * 2 + hi) * COUTP + nt * 16 + (lane & 15)) * 16;
      v8h b0 = *(const v8h*)(bp);
      v8h b1 = *(const v8h*)(bp + 8);
      v16h b;
#pragma unroll
      for (int i = 0; i < 8; ++i) { b[i] = b0[i]; b[8 + i] = b1[i]; }
      c = __builtin_amdgcn_wmma_f32_16x16x32_f16(false, a, false, b, (short)0, c,
                                                 false, false);
    }
    // C store: lane column ncol, rows m0+hi*8 .. +7 are contiguous positions.
    const int ncol = lane & 15;
    const int cout = nt * 16 + ncol;
    v8h sv;
#pragma unroll
    for (int r = 0; r < 8; ++r) sv[r] = (h16)c[r];
    if (COUTP == COUT || cout < COUT)
      *(v8h*)&syn[((size_t)n * COUT + cout) * (S * S) + m0 + hi * 8] = sv;
  }
}

// ---------------------------------------------------------------------------
// In-place IAF recurrence: one thread per f16 PAIR of (c,h,w); sequential T.
// b32 loads/stores, membrane accumulated in f32.
// ---------------------------------------------------------------------------
__global__ void iaf_kernel(h16* buf, int Bn, int T, int CHW) {
  const int CHW2 = CHW >> 1;
  size_t i = (size_t)blockIdx.x * blockDim.x + threadIdx.x;
  size_t total = (size_t)Bn * CHW2;
  if (i >= total) return;
  int b  = (int)(i / CHW2);
  int e2 = (int)(i - (size_t)b * CHW2);
  float v0 = 0.f, v1 = 0.f;
  for (int t = 0; t < T; ++t) {
    size_t idx = (size_t)(b * T + t) * CHW + 2 * e2;
    v2h d = *(const v2h*)(buf + idx);
    v0 += (float)d[0];
    v1 += (float)d[1];
    float s0 = (v0 >= THRESH) ? 1.f : 0.f;
    float s1 = (v1 >= THRESH) ? 1.f : 0.f;
    v0 -= THRESH * s0;
    v1 -= THRESH * s1;
    v2h o; o[0] = (h16)s0; o[1] = (h16)s1;
    *(v2h*)(buf + idx) = o;
  }
}

// 2x2 average pool (spike sums: exact in f16). v2h row-pair loads.
__global__ void pool_kernel(const h16* __restrict__ in, h16* __restrict__ out,
                            int N, int C, int S) {
  int SO = S >> 1;
  size_t total = (size_t)N * C * SO * SO;
  size_t i = (size_t)blockIdx.x * blockDim.x + threadIdx.x;
  if (i >= total) return;
  int x = (int)(i % SO); size_t t = i / SO;
  int y = (int)(t % SO); t /= SO;
  int c = (int)(t % C);
  int n = (int)(t / C);
  size_t base = ((size_t)n * C + c) * S * S + (size_t)(2 * y) * S + 2 * x;
  v2h p0 = *(const v2h*)(in + base);
  v2h p1 = *(const v2h*)(in + base + S);
  float s = (float)p0[0] + (float)p0[1] + (float)p1[0] + (float)p1[1];
  out[i] = (h16)(0.25f * s);
}

// ---------------------------------------------------------------------------
// Linear head: (1600 x 1024) @ (1024 x 16-pad) via WMMA; M=1600 = 25*64 exact,
// so no load guards. A/B fragments are b128 loads; out f32 (m,11).
// ---------------------------------------------------------------------------
__global__ __launch_bounds__(128)
void linear_wmma_kernel(const h16* __restrict__ act, const h16* __restrict__ wB,
                        float* __restrict__ out) {
  const int tid  = threadIdx.x;
  const int wave = tid >> 5;
  const int lane = tid & 31;
  const int hi   = lane >> 4;
  const int row  = blockIdx.x * 64 + wave * 16 + (lane & 15);
  const h16* arow = act + (size_t)row * 1024;
  v8f c = {};
#pragma unroll
  for (int kc = 0; kc < 32; ++kc) {
    v8h a0 = *(const v8h*)(arow + kc * 32 + hi * 16);
    v8h a1 = *(const v8h*)(arow + kc * 32 + hi * 16 + 8);
    v16h a;
#pragma unroll
    for (int i = 0; i < 8; ++i) { a[i] = a0[i]; a[8 + i] = a1[i]; }
    const h16* bp = wB + (size_t)((kc * 2 + hi) * 16 + (lane & 15)) * 16;
    v8h b0 = *(const v8h*)(bp);
    v8h b1 = *(const v8h*)(bp + 8);
    v16h b;
#pragma unroll
    for (int i = 0; i < 8; ++i) { b[i] = b0[i]; b[8 + i] = b1[i]; }
    c = __builtin_amdgcn_wmma_f32_16x16x32_f16(false, a, false, b, (short)0, c,
                                               false, false);
  }
  const int ncol = lane & 15;
#pragma unroll
  for (int r = 0; r < 8; ++r) {
    int m = blockIdx.x * 64 + wave * 16 + r + hi * 8;
    if (ncol < 11) out[(size_t)m * 11 + ncol] = c[r];
  }
}

// ---------------------------------------------------------------------------
extern "C" void kernel_launch(void* const* d_in, const int* in_sizes, int n_in,
                              void* d_out, int out_size, void* d_ws, size_t ws_size,
                              hipStream_t stream) {
  const float* x   = (const float*)d_in[0];
  const float* c0v = (const float*)d_in[1];  const float* c0g = (const float*)d_in[2];
  const float* c1v = (const float*)d_in[3];  const float* c1g = (const float*)d_in[4];
  const float* c2v = (const float*)d_in[5];  const float* c2g = (const float*)d_in[6];
  const float* c3v = (const float*)d_in[7];  const float* c3g = (const float*)d_in[8];
  const float* lv  = (const float*)d_in[9];  const float* lg  = (const float*)d_in[10];
  float* out = (float*)d_out;

  const int Bn = 32, T = 50, N = Bn * T;  // 1600 flattened images

  // Workspace layout (f16):
  //   bufA: 1600*8*64*64 halves = 104,857,600 B (syn/spike ping)
  //   bufB: 1600*8*32*32 halves =  26,214,400 B (pooled pong)
  //   packed f16 weights after that (~84 KB)
  char* ws   = (char*)d_ws;
  h16*  bufA = (h16*)ws;
  h16*  bufB = (h16*)(ws + 104857600);
  char* wb   = ws + 104857600 + 26214400;
  h16*  w0 = (h16*)(wb);
  h16*  w1 = (h16*)(wb + (size_t)(512) * 2);
  h16*  w2 = (h16*)(wb + (size_t)(512 + 1536) * 2);
  h16*  w3 = (h16*)(wb + (size_t)(512 + 1536 + 5120) * 2);
  h16*  wl = (h16*)(wb + (size_t)(512 + 1536 + 5120 + 18432) * 2);

  // Normalize + pack weights (f16, fragment-friendly layout).
  conv_wnorm_kernel<2, 8>  <<<16, 64, 0, stream>>>(c0v, c0g, w0);
  conv_wnorm_kernel<8, 16> <<<16, 64, 0, stream>>>(c1v, c1g, w1);
  conv_wnorm_kernel<16, 32><<<32, 64, 0, stream>>>(c2v, c2g, w2);
  conv_wnorm_kernel<32, 64><<<64, 64, 0, stream>>>(c3v, c3g, w3);
  lin_wnorm_kernel<<<16, 256, 0, stream>>>(lv, lg, wl);

  // Layer 0: conv(2->8, 64x64) -> IAF -> pool
  conv_wmma_kernel<2, 8, 64, float><<<dim3(N, 64), 128, 0, stream>>>(x, w0, bufA);
  { size_t tot = (size_t)Bn * 8 * 64 * 64 / 2;
    iaf_kernel<<<(unsigned)((tot + 255) / 256), 256, 0, stream>>>(bufA, Bn, T, 8 * 64 * 64); }
  { size_t tot = (size_t)N * 8 * 32 * 32;
    pool_kernel<<<(unsigned)((tot + 255) / 256), 256, 0, stream>>>(bufA, bufB, N, 8, 64); }

  // Layer 1: conv(8->16, 32x32) -> IAF -> pool
  conv_wmma_kernel<8, 16, 32, h16><<<dim3(N, 16), 128, 0, stream>>>(bufB, w1, bufA);
  { size_t tot = (size_t)Bn * 16 * 32 * 32 / 2;
    iaf_kernel<<<(unsigned)((tot + 255) / 256), 256, 0, stream>>>(bufA, Bn, T, 16 * 32 * 32); }
  { size_t tot = (size_t)N * 16 * 16 * 16;
    pool_kernel<<<(unsigned)((tot + 255) / 256), 256, 0, stream>>>(bufA, bufB, N, 16, 32); }

  // Layer 2: conv(16->32, 16x16) -> IAF -> pool
  conv_wmma_kernel<16, 32, 16, h16><<<dim3(N, 4), 128, 0, stream>>>(bufB, w2, bufA);
  { size_t tot = (size_t)Bn * 32 * 16 * 16 / 2;
    iaf_kernel<<<(unsigned)((tot + 255) / 256), 256, 0, stream>>>(bufA, Bn, T, 32 * 16 * 16); }
  { size_t tot = (size_t)N * 32 * 8 * 8;
    pool_kernel<<<(unsigned)((tot + 255) / 256), 256, 0, stream>>>(bufA, bufB, N, 32, 16); }

  // Layer 3: conv(32->64, 8x8) -> IAF -> pool
  conv_wmma_kernel<32, 64, 8, h16><<<dim3(N, 1), 128, 0, stream>>>(bufB, w3, bufA);
  { size_t tot = (size_t)Bn * 64 * 8 * 8 / 2;
    iaf_kernel<<<(unsigned)((tot + 255) / 256), 256, 0, stream>>>(bufA, Bn, T, 64 * 8 * 8); }
  { size_t tot = (size_t)N * 64 * 4 * 4;
    pool_kernel<<<(unsigned)((tot + 255) / 256), 256, 0, stream>>>(bufA, bufB, N, 64, 8); }

  // Linear head: (1600 x 1024) @ W^T -> (1600, 11)
  linear_wmma_kernel<<<25, 128, 0, stream>>>(bufB, wl, out);
}
